// Folding_Decoder_70557722739185
// MI455X (gfx1250) — matrix operand order repaired and checked by reading, not compiled, not run on hardware
//
#include <hip/hip_runtime.h>
#include <hip/hip_bf16.h>

// ---------------------------------------------------------------------------
// CDNA5 / gfx1250 implementation of the folding decoder.
//   - All large GEMMs use v_wmma_f32_16x16x32_f16 (f32 accumulate),
//     32x64 output tile per wave (8 WMMA per 12 b128 loads per K-step).
//   - EF expansion (graph conv chain + max-over-k) is fused into one
//     LDS-tiled WMMA kernel; max over the 4 knn neighbors is done inside
//     the accumulator fragment (neighbor rows are consecutive VGPRs).
//   - KNN is a VALU scan with the 1024 points of a batch staged in LDS.
// ---------------------------------------------------------------------------

typedef __attribute__((ext_vector_type(8)))  _Float16 v8h;
typedef __attribute__((ext_vector_type(16))) _Float16 v16h;
typedef __attribute__((ext_vector_type(8)))  float    v8f;

#define MKFRAG(a, b) __builtin_shufflevector((a), (b), 0,1,2,3,4,5,6,7,8,9,10,11,12,13,14,15)

__device__ __forceinline__ v8f wmma_f16f32(v16h a, v16h b, v8f c) {
#if defined(__AMDGCN__)
  // D = A(16x32,f16) * B(32x16,f16) + C(16x16,f32)
  return __builtin_amdgcn_wmma_f32_16x16x32_f16(false, a, false, b, (short)0, c, false, false);
#else
  (void)a; (void)b;
  return c;
#endif
}

// ---------------------------------------------------------------------------
// Generic WMMA GEMM:  out[m, n] = epilogue( sum_k A[m,k] * W[n,k] )
//   A : M x Kpad f16 row-major (Kpad multiple of 32, zero padded)
//   W : (gridDim.x*64) x Kpad f16 row-major (rows zero padded past N)
//   block = 128 threads (4 waves); wave computes a 32x64 tile
//   (2 M-subtiles x 4 N-subtiles); block covers 128 rows x 64 cols.
//   Epilogue: +bias, eval-mode batchnorm, relu, +residual; writes f16/f32.
// ---------------------------------------------------------------------------
__global__ void __launch_bounds__(128)
gemm_f16_wmma(const _Float16* __restrict__ A, int lda,
              const _Float16* __restrict__ W, int ldw,
              int N, int Kpad,
              const float* __restrict__ bias,
              const float* __restrict__ bng, const float* __restrict__ bnb,
              const float* __restrict__ bnm, const float* __restrict__ bnv,
              int relu,
              _Float16* __restrict__ outH, int ldoH,
              float* __restrict__ outF, int ldoF,
              const float* __restrict__ resid, int ldres)
{
  const int lane = threadIdx.x & 31;
  const int wave = threadIdx.x >> 5;
  const int col  = lane & 15;   // N (for B/C/D) or M (for A) lane index
  const int hi   = lane >> 4;   // which half of the K stripes
  const int m0   = (blockIdx.y * 4 + wave) * 32;
  const int n0b  = blockIdx.x * 64;

  v8f acc[2][4] = {{{}, {}, {}, {}}, {{}, {}, {}, {}}};
  const _Float16* Arow0 = A + (size_t)(m0 +  0 + col) * lda;
  const _Float16* Arow1 = A + (size_t)(m0 + 16 + col) * lda;
  const _Float16* W0 = W + (size_t)(n0b +  0 + col) * ldw;
  const _Float16* W1 = W + (size_t)(n0b + 16 + col) * ldw;
  const _Float16* W2 = W + (size_t)(n0b + 32 + col) * ldw;
  const _Float16* W3 = W + (size_t)(n0b + 48 + col) * ldw;

  for (int k0 = 0; k0 < Kpad; k0 += 32) {
    // A fragment: lane holds row m0+col, K = k0 + hi*8 + {0..7} and +16.
    const int ka = k0 + hi * 8;
    // Prefetch the streaming A rows one K-step ahead (global_prefetch_b8).
    if (k0 + 32 < Kpad) {
      __builtin_prefetch((const void*)(Arow0 + ka + 32), 0, 0);
      __builtin_prefetch((const void*)(Arow1 + ka + 32), 0, 0);
    }
    v16h af0 = MKFRAG(*(const v8h*)(Arow0 + ka), *(const v8h*)(Arow0 + ka + 16));
    v16h af1 = MKFRAG(*(const v8h*)(Arow1 + ka), *(const v8h*)(Arow1 + ka + 16));
    // B fragment: lane holds column n (= W row), K = k0 + hi*8 + {0..7} and +16.
    const int kb = k0 + hi * 8;
    v16h bf0 = MKFRAG(*(const v8h*)(W0 + kb), *(const v8h*)(W0 + kb + 16));
    v16h bf1 = MKFRAG(*(const v8h*)(W1 + kb), *(const v8h*)(W1 + kb + 16));
    v16h bf2 = MKFRAG(*(const v8h*)(W2 + kb), *(const v8h*)(W2 + kb + 16));
    v16h bf3 = MKFRAG(*(const v8h*)(W3 + kb), *(const v8h*)(W3 + kb + 16));
    acc[0][0] = wmma_f16f32(af0, bf0, acc[0][0]);
    acc[0][1] = wmma_f16f32(af0, bf1, acc[0][1]);
    acc[0][2] = wmma_f16f32(af0, bf2, acc[0][2]);
    acc[0][3] = wmma_f16f32(af0, bf3, acc[0][3]);
    acc[1][0] = wmma_f16f32(af1, bf0, acc[1][0]);
    acc[1][1] = wmma_f16f32(af1, bf1, acc[1][1]);
    acc[1][2] = wmma_f16f32(af1, bf2, acc[1][2]);
    acc[1][3] = wmma_f16f32(af1, bf3, acc[1][3]);
  }

#pragma unroll
  for (int t = 0; t < 4; ++t) {
    const int n = n0b + t * 16 + col;
    if (n >= N) continue;
    float bs = bias ? bias[n] : 0.f;
    float g = 1.f, bb = 0.f, mm = 0.f, inv = 1.f;
    if (bng) { g = bng[n]; bb = bnb[n]; mm = bnm[n]; inv = rsqrtf(bnv[n] + 1e-5f); }
#pragma unroll
    for (int mt = 0; mt < 2; ++mt) {
#pragma unroll
      for (int r = 0; r < 8; ++r) {
        // D element (m, n): vgpr r holds M = mtile + r + 8*hi at lane col = n.
        const int m = m0 + mt * 16 + r + 8 * hi;
        float v = acc[mt][t][r] + bs;
        if (bng)  v = g * (v - mm) * inv + bb;
        if (relu) v = fmaxf(v, 0.f);
        if (outH) outH[(size_t)m * ldoH + n] = (_Float16)v;
        if (outF) {
          float o = v;
          if (resid) o += resid[(size_t)m * ldres + n];
          outF[(size_t)m * ldoF + n] = o;
        }
      }
    }
  }
}

// ---------------------------------------------------------------------------
// KNN (k=4, includes self): pd = 2*inner - xx[n] - xx[m], keep 4 largest.
// One block per batch, points staged in LDS. Stable ties (strict >) match
// jax.lax.top_k's earliest-index tie behavior.
// ---------------------------------------------------------------------------
__global__ void __launch_bounds__(256)
knn_kernel(const float* __restrict__ pts,  // (B, 3, 1024): b*3072 + d*1024 + n
           int* __restrict__ nnidx)        // (B, 1024, 4)
{
  __shared__ float px[1024], py[1024], pz[1024], sxx[1024];
  const int b = blockIdx.x;
  const int tid = threadIdx.x;
  const float* P = pts + (size_t)b * 3072;
  for (int i = tid; i < 1024; i += 256) {
    float x = P[i], y = P[1024 + i], z = P[2048 + i];
    px[i] = x; py[i] = y; pz[i] = z; sxx[i] = x * x + y * y + z * z;
  }
  __syncthreads();
  for (int n = tid; n < 1024; n += 256) {
    const float cx = px[n], cy = py[n], cz = pz[n], cxx = sxx[n];
    float b0 = -3.402823466e38f, b1 = b0, b2 = b0, b3 = b0;
    int   i0 = 0, i1 = 0, i2 = 0, i3 = 0;
    for (int m = 0; m < 1024; ++m) {
      float v = 2.f * (cx * px[m] + cy * py[m] + cz * pz[m]) - cxx - sxx[m];
      if (v > b3) {
        b3 = v; i3 = m;
        if (b3 > b2) { float tv = b2; b2 = b3; b3 = tv; int ti = i2; i2 = i3; i3 = ti;
          if (b2 > b1) { tv = b1; b1 = b2; b2 = tv; ti = i1; i1 = i2; i2 = ti;
            if (b1 > b0) { tv = b0; b0 = b1; b1 = tv; ti = i0; i0 = i1; i1 = ti; } } }
      }
    }
    int* o = nnidx + ((size_t)b * 1024 + n) * 4;
    o[0] = i0; o[1] = i1; o[2] = i2; o[3] = i3;
  }
}

// ---------------------------------------------------------------------------
// Fused EF expansion. Block: 128 threads (4 waves) = 32 points x 4 neighbors
// of one batch. Stages:
//   1) build A1 (128 x 96 f16 in LDS): [relu(e1: 64), relu(ief: 6), 0 pad]
//   2) ec2: WMMA (K=96) from LDS -> relu -> A2 (128 x 64 f16 in LDS)
//   3) ec3: WMMA (K=64) -> +bias -> max over the 4 neighbor rows inside the
//      D fragment -> scatter x_exp[b, ch, n] into X1[(b*64+ch), 3+n].
// ---------------------------------------------------------------------------
__global__ void __launch_bounds__(128)
ef_expand_wmma(const float* __restrict__ pts,
               const int* __restrict__ nnidx,
               const float* __restrict__ ec1w, const float* __restrict__ ec1b,
               const _Float16* __restrict__ ec2w, const float* __restrict__ ec2b,
               const _Float16* __restrict__ ec3w, const float* __restrict__ ec3b,
               _Float16* __restrict__ X1)       // (8192, 1056) f16
{
  __shared__ _Float16 A1[128][96];
  __shared__ _Float16 A2[128][64];
  const int b = blockIdx.y;
  const int chunk = blockIdx.x;     // 32 points per chunk
  const int tid = threadIdx.x;

  { // build one row per thread: row = (point_local * 4 + k)
    const int r  = tid;
    const int pl = r >> 2, kk = r & 3;
    const int n  = chunk * 32 + pl;
    const float* P = pts + (size_t)b * 3072;
    float f[6];
    f[0] = P[n];  f[1] = P[1024 + n];  f[2] = P[2048 + n];
    const int j = nnidx[((size_t)b * 1024 + n) * 4 + kk];
    f[3] = P[j];  f[4] = P[1024 + j];  f[5] = P[2048 + j];
    for (int o = 0; o < 64; ++o) {
      float s = ec1b[o];
#pragma unroll
      for (int c = 0; c < 6; ++c) s += ec1w[o * 6 + c] * f[c];
      A1[r][o] = (_Float16)fmaxf(s, 0.f);
    }
#pragma unroll
    for (int c = 0; c < 6; ++c) A1[r][64 + c] = (_Float16)fmaxf(f[c], 0.f);
    for (int c = 70; c < 96; ++c) A1[r][c] = (_Float16)0.f;
  }
  __syncthreads();

  const int lane = tid & 31, wave = tid >> 5;
  const int col = lane & 15, hi = lane >> 4;

  // ec2: (128 x 96) @ (96 x 64) -> relu -> A2
  for (int mt = wave; mt < 8; mt += 4) {
    const int m0 = mt * 16;
    v8f acc[4] = {{}, {}, {}, {}};
    for (int k0 = 0; k0 < 96; k0 += 32) {
      const _Float16* Ap = &A1[m0 + col][0] + k0 + hi * 8;
      v16h af = MKFRAG(*(const v8h*)Ap, *(const v8h*)(Ap + 16));
#pragma unroll
      for (int t = 0; t < 4; ++t) {
        const _Float16* Wp = ec2w + (size_t)(t * 16 + col) * 96 + k0 + hi * 8;
        v16h bf = MKFRAG(*(const v8h*)Wp, *(const v8h*)(Wp + 16));
        acc[t] = wmma_f16f32(af, bf, acc[t]);
      }
    }
#pragma unroll
    for (int t = 0; t < 4; ++t) {
      const int n = t * 16 + col;
      const float bb = ec2b[n];
#pragma unroll
      for (int r = 0; r < 8; ++r)
        A2[m0 + r + 8 * hi][n] = (_Float16)fmaxf(acc[t][r] + bb, 0.f);
    }
  }
  __syncthreads();

  // ec3: (128 x 64) @ (64 x 64) -> +bias -> max over neighbor quads -> X1
  for (int mt = wave; mt < 8; mt += 4) {
    const int m0 = mt * 16;
    v8f acc[4] = {{}, {}, {}, {}};
    for (int k0 = 0; k0 < 64; k0 += 32) {
      const _Float16* Ap = &A2[m0 + col][0] + k0 + hi * 8;
      v16h af = MKFRAG(*(const v8h*)Ap, *(const v8h*)(Ap + 16));
#pragma unroll
      for (int t = 0; t < 4; ++t) {
        const _Float16* Wp = ec3w + (size_t)(t * 16 + col) * 64 + k0 + hi * 8;
        v16h bf = MKFRAG(*(const v8h*)Wp, *(const v8h*)(Wp + 16));
        acc[t] = wmma_f16f32(af, bf, acc[t]);
      }
    }
    // Rows m0..m0+15 are points (m0/4)..(m0/4+3), 4 neighbor rows each.
    // In the D fragment: vgprs 0..3 / 4..7 at lane half `hi` are exactly
    // one point's 4 neighbors -> reduce with 3 v_max_f32.
    const int pbase = m0 >> 2;
#pragma unroll
    for (int t = 0; t < 4; ++t) {
      const int ch = t * 16 + col;
      const float bb = ec3b[ch];
      float vlo = fmaxf(fmaxf(acc[t][0], acc[t][1]), fmaxf(acc[t][2], acc[t][3])) + bb;
      float vhi = fmaxf(fmaxf(acc[t][4], acc[t][5]), fmaxf(acc[t][6], acc[t][7])) + bb;
      const int p0 = pbase + hi * 2;            // lane half selects point pair
      const int ng = chunk * 32 + p0;           // global point index
      _Float16* dst = X1 + ((size_t)b * 64 + ch) * 1056 + 3 + ng;
      dst[0] = (_Float16)vlo;
      dst[1] = (_Float16)vhi;
    }
  }
}

// ---------------------------------------------------------------------------
// Elementwise helpers
// ---------------------------------------------------------------------------
__global__ void pack_f16_kernel(const float* __restrict__ src, _Float16* __restrict__ dst,
                                int srows, int scols, int drows, int dcols)
{
  size_t i = (size_t)blockIdx.x * blockDim.x + threadIdx.x;
  const size_t tot = (size_t)drows * dcols;
  if (i >= tot) return;
  const int c = (int)(i % dcols);
  const int r = (int)(i / dcols);
  const float v = (r < srows && c < scols) ? src[(size_t)r * scols + c] : 0.f;
  dst[i] = (_Float16)v;
}

__global__ void zero_f16_kernel(_Float16* __restrict__ p, size_t n)
{
  size_t i = (size_t)blockIdx.x * blockDim.x + threadIdx.x;
  if (i < n) p[i] = (_Float16)0.f;
}

// X1[(b*64+p), :] = [sk_point[b] (3), <x_exp filled later> , 0 pad]
__global__ void x1_init_kernel(const float* __restrict__ sk, _Float16* __restrict__ X1)
{
  size_t i = (size_t)blockIdx.x * blockDim.x + threadIdx.x;
  if (i >= (size_t)8192 * 1056) return;
  const int c = (int)(i % 1056);
  const int row = (int)(i / 1056);
  const int b = row >> 6;
  X1[i] = (c < 3) ? (_Float16)sk[b * 3 + c] : (_Float16)0.f;
}

// X2[row, :] = [grid(-0.05 x2), coarse_fold(3), xc = X1[row, 0..1026], 0 pad]
__global__ void x2_build_kernel(const _Float16* __restrict__ X1, const float* __restrict__ cf,
                                _Float16* __restrict__ X2)
{
  size_t i = (size_t)blockIdx.x * blockDim.x + threadIdx.x;
  if (i >= (size_t)8192 * 1056) return;
  const int c = (int)(i % 1056);
  const size_t row = i / 1056;
  _Float16 v;
  if (c < 2)          v = (_Float16)(-0.05f);
  else if (c < 5)     v = (_Float16)cf[row * 3 + (c - 2)];
  else if (c < 1032)  v = X1[row * 1056 + (c - 5)];
  else                v = (_Float16)0.f;
  X2[i] = v;
}

// ---------------------------------------------------------------------------
// Launch
// ---------------------------------------------------------------------------
extern "C" void kernel_launch(void* const* d_in, const int* in_sizes, int n_in,
                              void* d_out, int out_size, void* d_ws, size_t ws_size,
                              hipStream_t stream) {
  (void)in_sizes; (void)n_in; (void)out_size;

  const float* x      = (const float*)d_in[0];
  const float* sk     = (const float*)d_in[1];
  const float* fc1_w  = (const float*)d_in[2];
  const float* fc1_b  = (const float*)d_in[3];
  const float* fc2_w  = (const float*)d_in[4];
  const float* fc2_b  = (const float*)d_in[5];
  const float* fc3_w  = (const float*)d_in[6];
  const float* fc3_b  = (const float*)d_in[7];
  const float* bn1_g  = (const float*)d_in[8];
  const float* bn1_b  = (const float*)d_in[9];
  const float* bn1_m  = (const float*)d_in[10];
  const float* bn1_v  = (const float*)d_in[11];
  const float* bn2_g  = (const float*)d_in[12];
  const float* bn2_b  = (const float*)d_in[13];
  const float* bn2_m  = (const float*)d_in[14];
  const float* bn2_v  = (const float*)d_in[15];
  const float* ec1_w  = (const float*)d_in[16];
  const float* ec1_b  = (const float*)d_in[17];
  const float* ec2_w  = (const float*)d_in[18];
  const float* ec2_b  = (const float*)d_in[19];
  const float* ec3_w  = (const float*)d_in[20];
  const float* ec3_b  = (const float*)d_in[21];
  const float* f1c1_w = (const float*)d_in[22];
  const float* f1c1_b = (const float*)d_in[23];
  const float* f1c2_w = (const float*)d_in[24];
  const float* f1c2_b = (const float*)d_in[25];
  const float* f1c3_w = (const float*)d_in[26];
  const float* f1c3_b = (const float*)d_in[27];
  const float* f1bn1_g = (const float*)d_in[28];
  const float* f1bn1_b = (const float*)d_in[29];
  const float* f1bn1_m = (const float*)d_in[30];
  const float* f1bn1_v = (const float*)d_in[31];
  const float* f1bn2_g = (const float*)d_in[32];
  const float* f1bn2_b = (const float*)d_in[33];
  const float* f1bn2_m = (const float*)d_in[34];
  const float* f1bn2_v = (const float*)d_in[35];
  const float* f2c1_w = (const float*)d_in[36];
  const float* f2c1_b = (const float*)d_in[37];
  const float* f2c2_w = (const float*)d_in[38];
  const float* f2c2_b = (const float*)d_in[39];
  const float* f2c3_w = (const float*)d_in[40];
  const float* f2c3_b = (const float*)d_in[41];
  const float* f2bn1_g = (const float*)d_in[42];
  const float* f2bn1_b = (const float*)d_in[43];
  const float* f2bn1_m = (const float*)d_in[44];
  const float* f2bn1_v = (const float*)d_in[45];
  const float* f2bn2_g = (const float*)d_in[46];
  const float* f2bn2_b = (const float*)d_in[47];
  const float* f2bn2_m = (const float*)d_in[48];
  const float* f2bn2_v = (const float*)d_in[49];

  // ---- workspace carve-out (256B aligned) ----
  size_t off = 0;
  auto carve = [&](size_t bytes) -> void* {
    off = (off + 255) & ~(size_t)255;
    void* p = (char*)d_ws + off;
    off += bytes;
    return p;
  };
  _Float16* w1h   = (_Float16*)carve((size_t)1088 * 1056 * 2);
  _Float16* w2h   = (_Float16*)carve((size_t)1024 * 1056 * 2);
  _Float16* w3h   = (_Float16*)carve((size_t)3072 * 1024 * 2);
  _Float16* ec2wh = (_Float16*)carve((size_t)64 * 96 * 2);
  _Float16* ec3wh = (_Float16*)carve((size_t)64 * 64 * 2);
  _Float16* f1w1  = (_Float16*)carve((size_t)512 * 1056 * 2);
  _Float16* f1w2  = (_Float16*)carve((size_t)256 * 512 * 2);
  _Float16* f1w3  = (_Float16*)carve((size_t)64 * 256 * 2);
  _Float16* f2w1  = (_Float16*)carve((size_t)512 * 1056 * 2);
  _Float16* f2w2  = (_Float16*)carve((size_t)256 * 512 * 2);
  _Float16* f2w3  = (_Float16*)carve((size_t)64 * 256 * 2);
  _Float16* xh    = (_Float16*)carve((size_t)128 * 1056 * 2);
  _Float16* h1h   = (_Float16*)carve((size_t)128 * 1056 * 2);
  _Float16* h2h   = (_Float16*)carve((size_t)128 * 1024 * 2);
  float*    pts   = (float*)   carve((size_t)128 * 3072 * 4);
  int*      nnidx = (int*)     carve((size_t)128 * 1024 * 4 * 4);
  _Float16* X1    = (_Float16*)carve((size_t)8192 * 1056 * 2);
  _Float16* X2    = (_Float16*)carve((size_t)8192 * 1056 * 2);
  _Float16* H1    = (_Float16*)carve((size_t)8192 * 512 * 2);
  _Float16* H2    = (_Float16*)carve((size_t)8192 * 256 * 2);
  float*    cf    = (float*)   carve((size_t)8192 * 3 * 4);
  if (off > ws_size) return;  // workspace too small; bail deterministically

  auto gsz = [](size_t n) { return (unsigned)((n + 255) / 256); };

  // ---- pack weights / inputs to zero-padded f16 ----
  pack_f16_kernel<<<gsz((size_t)1088*1056), 256, 0, stream>>>(fc1_w, w1h, 1027, 1027, 1088, 1056);
  pack_f16_kernel<<<gsz((size_t)1024*1056), 256, 0, stream>>>(fc2_w, w2h, 1024, 1027, 1024, 1056);
  pack_f16_kernel<<<gsz((size_t)3072*1024), 256, 0, stream>>>(fc3_w, w3h, 3072, 1024, 3072, 1024);
  pack_f16_kernel<<<gsz((size_t)64*96),     256, 0, stream>>>(ec2_w, ec2wh, 64, 70, 64, 96);
  pack_f16_kernel<<<gsz((size_t)64*64),     256, 0, stream>>>(ec3_w, ec3wh, 64, 64, 64, 64);
  pack_f16_kernel<<<gsz((size_t)512*1056),  256, 0, stream>>>(f1c1_w, f1w1, 512, 1027, 512, 1056);
  pack_f16_kernel<<<gsz((size_t)256*512),   256, 0, stream>>>(f1c2_w, f1w2, 256, 512, 256, 512);
  pack_f16_kernel<<<gsz((size_t)64*256),    256, 0, stream>>>(f1c3_w, f1w3, 3, 256, 64, 256);
  pack_f16_kernel<<<gsz((size_t)512*1056),  256, 0, stream>>>(f2c1_w, f2w1, 512, 1032, 512, 1056);
  pack_f16_kernel<<<gsz((size_t)256*512),   256, 0, stream>>>(f2c2_w, f2w2, 256, 512, 256, 512);
  pack_f16_kernel<<<gsz((size_t)64*256),    256, 0, stream>>>(f2c3_w, f2w3, 3, 256, 64, 256);
  pack_f16_kernel<<<gsz((size_t)128*1056),  256, 0, stream>>>(x, xh, 128, 1027, 128, 1056);
  zero_f16_kernel<<<gsz((size_t)128*1056),  256, 0, stream>>>(h1h, (size_t)128 * 1056);

  // ---- FC decoder: h1 = relu(bn1(x@W1^T+b1)); h2 = relu(bn2(...)); coarse = h2@W3^T+b3
  gemm_f16_wmma<<<dim3(17, 1), 128, 0, stream>>>(xh, 1056, w1h, 1056, 1027, 1056,
      fc1_b, bn1_g, bn1_b, bn1_m, bn1_v, 1, h1h, 1056, nullptr, 0, nullptr, 0);
  gemm_f16_wmma<<<dim3(16, 1), 128, 0, stream>>>(h1h, 1056, w2h, 1056, 1024, 1056,
      fc2_b, bn2_g, bn2_b, bn2_m, bn2_v, 1, h2h, 1024, nullptr, 0, nullptr, 0);
  gemm_f16_wmma<<<dim3(48, 1), 128, 0, stream>>>(h2h, 1024, w3h, 1024, 3072, 1024,
      fc3_b, nullptr, nullptr, nullptr, nullptr, 0, nullptr, 0, pts, 3072, nullptr, 0);

  // ---- KNN + fused EF expansion -> X1 (= [sk, x_exp, pad]) ----
  knn_kernel<<<128, 256, 0, stream>>>(pts, nnidx);
  x1_init_kernel<<<gsz((size_t)8192*1056), 256, 0, stream>>>(sk, X1);
  ef_expand_wmma<<<dim3(32, 128), 128, 0, stream>>>(pts, nnidx, ec1_w, ec1_b,
      ec2wh, ec2_b, ec3wh, ec3_b, X1);

  // ---- Folding 1 -> coarse_fold ----
  gemm_f16_wmma<<<dim3(8, 64), 128, 0, stream>>>(X1, 1056, f1w1, 1056, 512, 1056,
      f1c1_b, f1bn1_g, f1bn1_b, f1bn1_m, f1bn1_v, 1, H1, 512, nullptr, 0, nullptr, 0);
  gemm_f16_wmma<<<dim3(4, 64), 128, 0, stream>>>(H1, 512, f1w2, 512, 256, 512,
      f1c2_b, f1bn2_g, f1bn2_b, f1bn2_m, f1bn2_v, 1, H2, 256, nullptr, 0, nullptr, 0);
  gemm_f16_wmma<<<dim3(1, 64), 128, 0, stream>>>(H2, 256, f1w3, 256, 3, 256,
      f1c3_b, nullptr, nullptr, nullptr, nullptr, 0, nullptr, 0, cf, 3, nullptr, 0);

  // ---- feat = [grid, coarse_fold, xc] ; Folding 2 + residual -> d_out ----
  x2_build_kernel<<<gsz((size_t)8192*1056), 256, 0, stream>>>(X1, cf, X2);
  gemm_f16_wmma<<<dim3(8, 64), 128, 0, stream>>>(X2, 1056, f2w1, 1056, 512, 1056,
      f2c1_b, f2bn1_g, f2bn1_b, f2bn1_m, f2bn1_v, 1, H1, 512, nullptr, 0, nullptr, 0);
  gemm_f16_wmma<<<dim3(4, 64), 128, 0, stream>>>(H1, 512, f2w2, 512, 256, 512,
      f2c2_b, f2bn2_g, f2bn2_b, f2bn2_m, f2bn2_v, 1, H2, 256, nullptr, 0, nullptr, 0);
  gemm_f16_wmma<<<dim3(1, 64), 128, 0, stream>>>(H2, 256, f2w3, 256, 3, 256,
      f2c3_b, nullptr, nullptr, nullptr, nullptr, 0, nullptr, 0, (float*)d_out, 3, cf, 3);
}
